// NNARX_42262478193095
// MI455X (gfx1250) — compile-verified, matrix operand order checked by reading, not run on hardware
//
#include <hip/hip_runtime.h>

#define T_STEPS 1024
#define BATCH   1024
#define INU     32
#define INY     8
#define HDIM    512
#define OUTD    8

typedef __attribute__((ext_vector_type(16))) __bf16        v16bf;
typedef __attribute__((ext_vector_type(8)))  float         v8f;
typedef __attribute__((ext_vector_type(4)))  unsigned int  u32x4;

union AFrag { v16bf v; u32x4 q[2]; };

__device__ __forceinline__ unsigned short f2bf_bits(float f) {
    union { float f; unsigned u; } x; x.f = f;
    unsigned r = x.u + 0x7FFFu + ((x.u >> 16) & 1u);   // round-to-nearest-even
    return (unsigned short)(r >> 16);
}
__device__ __forceinline__ float bf2f(unsigned short h) {
    union { unsigned u; float f; } x; x.u = ((unsigned)h) << 16;
    return x.f;
}

__device__ __forceinline__ v8f wmma_bf16(v16bf a, v16bf b, v8f c) {
    return __builtin_amdgcn_wmma_f32_16x16x32_bf16(
        /*neg_a=*/false, a, /*neg_b=*/false, b,
        /*c_mod=*/(short)0, c, /*reuse_a=*/false, /*reuse_b=*/false);
}

// A-fragment (16x32 bf16, MxK) from LDS row-major [16][stride] bf16 buffer.
// Lane l: row = l%16, k0 = kt*32 + (l/16)*8; elems 0..7 = K k0..k0+7,
// elems 8..15 = K k0+16..k0+23  -> two 16B ds loads.
__device__ __forceinline__ v16bf load_a(const unsigned short* p, int stride,
                                        int kt, int lane) {
    int m  = lane & 15;
    int k0 = kt * 32 + ((lane >> 4) << 3);
    const unsigned short* base = p + m * stride + k0;
    AFrag a;
    a.q[0] = *(const u32x4*)(base);
    a.q[1] = *(const u32x4*)(base + 16);
    return a.v;
}

// B-fragment: pre-packed tile of 512 halves in global ws; lane reads its 32B.
__device__ __forceinline__ v16bf load_b(const unsigned short* __restrict__ g,
                                        int lane) {
    const u32x4* p = (const u32x4*)(g + lane * 16);
    AFrag b;
    b.q[0] = p[0];
    b.q[1] = p[1];
    return b.v;
}

// Pack W (row-major [R,C]) as B = W^T in WMMA bf16 B-fragment tile layout.
// Tile (kt,nt) covers K = kt*32.., N = nt*16..; slot s = lane*16+e holds
// B[k][n] with n = nt*16 + lane%16, k = kt*32 + (lane/16)*8 + (e<8 ? e : e+8).
__global__ void nnarx_pack(const float* __restrict__ W, int R, int C,
                           int ntiles, int total,
                           unsigned short* __restrict__ dst) {
    int idx = blockIdx.x * blockDim.x + threadIdx.x;
    if (idx >= total) return;
    int tile = idx >> 9;
    int s    = idx & 511;
    int kt   = tile / ntiles;
    int nt   = tile - kt * ntiles;
    int lane = s >> 4;
    int e    = s & 15;
    int n = nt * 16 + (lane & 15);
    int k = kt * 32 + ((lane >> 4) << 3) + (e < 8 ? e : e + 8);
    float v = (n < R && k < C) ? W[n * C + k] : 0.0f;
    dst[idx] = f2bf_bits(v);
}

__global__ void __launch_bounds__(256)
nnarx_main(const float* __restrict__ u_seq, const float* __restrict__ y_init,
           const float* __restrict__ b_ih,  const float* __restrict__ b_hh,
           const float* __restrict__ b1,    const float* __restrict__ b2,
           const unsigned short* __restrict__ wih,  // [2*96*512]  K=64  N=1536
           const unsigned short* __restrict__ whh,  // [16*96*512] K=512 N=1536
           const unsigned short* __restrict__ w1,   // [16*32*512] K=512 N=512
           const unsigned short* __restrict__ w2,   // [16*1*512]  K=512 N=16
           float* __restrict__ out) {
    __shared__ __align__(16) unsigned short hbuf0[16 * HDIM];
    __shared__ __align__(16) unsigned short hbuf1[16 * HDIM];
    __shared__ __align__(16) unsigned short hidb [16 * HDIM];
    __shared__ __align__(16) unsigned short xbuf [16 * 64];
    __shared__ float ybuf[16 * OUTD];

    const int tid   = threadIdx.x;
    const int lane  = tid & 31;
    const int wave  = tid >> 5;              // 0..7
    const int bBase = blockIdx.x * 16;       // 16-row batch tile per block

    // init: h = 0, y = y_init
    for (int i = tid; i < 16 * HDIM; i += 256) hbuf0[i] = 0;
    for (int i = tid; i < 16 * OUTD; i += 256) {
        int m = i >> 3, o = i & 7;
        ybuf[i] = y_init[(bBase + m) * OUTD + o];
    }
    __syncthreads();

    int parity = 0;
#pragma unroll 1
    for (int t = 0; t < T_STEPS; ++t) {
        unsigned short* hc = parity ? hbuf1 : hbuf0;  // current h
        unsigned short* hx = parity ? hbuf0 : hbuf1;  // next h

        // ---- phase 1: stage x = [u_t | y | 0] as bf16 [16][64] ----
        for (int i = tid; i < 16 * 64; i += 256) {
            int m = i >> 6, c = i & 63;
            float v = 0.0f;
            if (c < INU)
                v = u_seq[((size_t)(bBase + m) * T_STEPS + t) * INU + c];
            else if (c < INU + INY)
                v = ybuf[m * OUTD + (c - INU)];
            xbuf[i] = f2bf_bits(v);
        }
        if (t + 1 < T_STEPS && tid < 16)   // global_prefetch_b8 next step's u
            __builtin_prefetch(
                &u_seq[((size_t)(bBase + tid) * T_STEPS + (t + 1)) * INU], 0, 0);
        __syncthreads();

        // ---- phase 2: GRU gates; each wave owns 4 hidden N-tiles ----
#pragma unroll 1
        for (int jj = 0; jj < 4; ++jj) {
            int jt = wave * 4 + jj;                  // hidden tile 0..31
            v8f accr = {}, accz = {}, accxn = {}, acchn = {};
            // gx: K = 64 (x padded), N-tiles jt / 32+jt / 64+jt  (r|z|n)
#pragma unroll
            for (int kt = 0; kt < 2; ++kt) {
                v16bf a  = load_a(xbuf, 64, kt, lane);
                v16bf br = load_b(wih + ((size_t)(kt * 96 + jt)      << 9), lane);
                v16bf bz = load_b(wih + ((size_t)(kt * 96 + 32 + jt) << 9), lane);
                v16bf bn = load_b(wih + ((size_t)(kt * 96 + 64 + jt) << 9), lane);
                accr  = wmma_bf16(a, br, accr);
                accz  = wmma_bf16(a, bz, accz);
                accxn = wmma_bf16(a, bn, accxn);
            }
            // gh: K = 512
#pragma unroll 4
            for (int kt = 0; kt < 16; ++kt) {
                v16bf a  = load_a(hc, HDIM, kt, lane);
                v16bf br = load_b(whh + ((size_t)(kt * 96 + jt)      << 9), lane);
                v16bf bz = load_b(whh + ((size_t)(kt * 96 + 32 + jt) << 9), lane);
                v16bf bn = load_b(whh + ((size_t)(kt * 96 + 64 + jt) << 9), lane);
                accr  = wmma_bf16(a, br, accr);
                accz  = wmma_bf16(a, bz, accz);
                acchn = wmma_bf16(a, bn, acchn);
            }
            // elementwise gates on D layout: lane -> col n = lane%16,
            // VGPR v -> row m = v + (lane/16)*8
            int nn = lane & 15;
            int j  = jt * 16 + nn;
            float bir   = b_ih[j]            + b_hh[j];
            float biz   = b_ih[HDIM + j]     + b_hh[HDIM + j];
            float bin_x = b_ih[2 * HDIM + j];
            float bin_h = b_hh[2 * HDIM + j];
            int mBase = (lane >> 4) * 8;
#pragma unroll
            for (int v = 0; v < 8; ++v) {
                int m = mBase + v;
                float r  = 1.0f / (1.0f + __expf(-(accr[v] + bir)));
                float z  = 1.0f / (1.0f + __expf(-(accz[v] + biz)));
                float ng = tanhf((accxn[v] + bin_x) + r * (acchn[v] + bin_h));
                float ho = bf2f(hc[m * HDIM + j]);
                float hv = (1.0f - z) * ng + z * ho;
                hx[m * HDIM + j] = f2bf_bits(hv);
            }
        }
        __syncthreads();

        // ---- phase 3: hid = relu(h_new @ W1^T + b1) ----
#pragma unroll 1
        for (int jj = 0; jj < 4; ++jj) {
            int nt = wave * 4 + jj;                  // 0..31
            v8f acc = {};
#pragma unroll 4
            for (int kt = 0; kt < 16; ++kt) {
                v16bf a = load_a(hx, HDIM, kt, lane);
                v16bf b = load_b(w1 + ((size_t)(kt * 32 + nt) << 9), lane);
                acc = wmma_bf16(a, b, acc);
            }
            int nn  = lane & 15;
            int col = nt * 16 + nn;
            float bb = b1[col];
            int mBase = (lane >> 4) * 8;
#pragma unroll
            for (int v = 0; v < 8; ++v) {
                float hv = acc[v] + bb;
                hv = hv > 0.0f ? hv : 0.0f;
                hidb[(mBase + v) * HDIM + col] = f2bf_bits(hv);
            }
        }
        __syncthreads();

        // ---- phase 4: pred = hid @ W2^T + b2 (N padded 8->16; wave 0) ----
        if (wave == 0) {
            v8f acc = {};
#pragma unroll 4
            for (int kt = 0; kt < 16; ++kt) {
                v16bf a = load_a(hidb, HDIM, kt, lane);
                v16bf b = load_b(w2 + ((size_t)kt << 9), lane);
                acc = wmma_bf16(a, b, acc);
            }
            int nn = lane & 15;
            int mBase = (lane >> 4) * 8;
            if (nn < OUTD) {
                float bb = b2[nn];
#pragma unroll
                for (int v = 0; v < 8; ++v) {
                    int m = mBase + v;
                    float p = acc[v] + bb;
                    out[((size_t)(bBase + m) * T_STEPS + t) * OUTD + nn] = p;
                    ybuf[m * OUTD + nn] = p;   // feedback for next step's x
                }
            }
        }
        __syncthreads();
        parity ^= 1;
    }
}

extern "C" void kernel_launch(void* const* d_in, const int* in_sizes, int n_in,
                              void* d_out, int out_size, void* d_ws, size_t ws_size,
                              hipStream_t stream) {
    const float* u_seq  = (const float*)d_in[0];
    const float* y_init = (const float*)d_in[1];
    const float* W_ih   = (const float*)d_in[2];
    const float* W_hh   = (const float*)d_in[3];
    const float* b_ih   = (const float*)d_in[4];
    const float* b_hh   = (const float*)d_in[5];
    const float* W1     = (const float*)d_in[6];
    const float* b1     = (const float*)d_in[7];
    const float* W2     = (const float*)d_in[8];
    const float* b2     = (const float*)d_in[9];
    unsigned short* ws  = (unsigned short*)d_ws;

    const size_t n_wih = (size_t)2  * 96 * 512;   // K=64  N=1536
    const size_t n_whh = (size_t)16 * 96 * 512;   // K=512 N=1536
    const size_t n_w1  = (size_t)16 * 32 * 512;   // K=512 N=512
    const size_t n_w2  = (size_t)16 * 1  * 512;   // K=512 N=16
    unsigned short* p_wih = ws;
    unsigned short* p_whh = p_wih + n_wih;
    unsigned short* p_w1  = p_whh + n_whh;
    unsigned short* p_w2  = p_w1  + n_w1;

    nnarx_pack<<<(int)((n_wih + 255) / 256), 256, 0, stream>>>(W_ih, 1536, 40,  96, (int)n_wih, p_wih);
    nnarx_pack<<<(int)((n_whh + 255) / 256), 256, 0, stream>>>(W_hh, 1536, 512, 96, (int)n_whh, p_whh);
    nnarx_pack<<<(int)((n_w1  + 255) / 256), 256, 0, stream>>>(W1,   512,  512, 32, (int)n_w1,  p_w1);
    nnarx_pack<<<(int)((n_w2  + 255) / 256), 256, 0, stream>>>(W2,   8,    512, 1,  (int)n_w2,  p_w2);

    nnarx_main<<<BATCH / 16, 256, 0, stream>>>(
        u_seq, y_init, b_ih, b_hh, b1, b2,
        p_wih, p_whh, p_w1, p_w2, (float*)d_out);
}
